// BertSelfAttention_27676769256167
// MI455X (gfx1250) — matrix operand order repaired and checked by reading, not compiled, not run on hardware
//
#include <hip/hip_runtime.h>

// BERT self-attention for MI455X (gfx1250, wave32, WMMA bf16).
// B=4, S=2048, H=1024, heads=16, dh=64. fp32 I/O, bf16 matrix-core math
// with fp32 accumulation (v_wmma_f32_16x16x32_bf16).

#define B_   4
#define S_   2048
#define H_   1024
#define NH_  16
#define DH_  64

typedef __attribute__((ext_vector_type(16))) __bf16          v16bf;
typedef __attribute__((ext_vector_type(8)))  float           v8f;
typedef __attribute__((ext_vector_type(8)))  unsigned short  ushort8;
typedef __attribute__((ext_vector_type(16))) unsigned short  ushort16;

__device__ __forceinline__ unsigned short f2bf(float f) {
  unsigned int u = __float_as_uint(f);
  u += 0x7FFFu + ((u >> 16) & 1u);          // round-to-nearest-even
  return (unsigned short)(u >> 16);
}

__device__ __forceinline__ v8f wmma_bf16(v16bf a, v16bf b, v8f c) {
  // 8 args: (neg_a, A, neg_b, B, c_mod, C, reuse_a, reuse_b)
  return __builtin_amdgcn_wmma_f32_16x16x32_bf16(false, a, false, b,
                                                 (short)0, c, false, false);
}

// Build a 16-element bf16 fragment from two contiguous 8-element LDS chunks.
__device__ __forceinline__ v16bf frag2(const unsigned short* p0,
                                       const unsigned short* p1) {
  ushort8 a = *(const ushort8*)p0;
  ushort8 b = *(const ushort8*)p1;
  ushort16 r = __builtin_shufflevector(a, b, 0,1,2,3,4,5,6,7,
                                             8,9,10,11,12,13,14,15);
  return __builtin_bit_cast(v16bf, r);
}

// ---------------------------------------------------------------------------
// Kernel 1: Y = X @ W^T + b  -> bf16 workspace in [B, h, S, dh] layout.
// grid = (N/64, M/128, 3) ; block = 256 (8 wave32)
// ---------------------------------------------------------------------------
__global__ __launch_bounds__(256)
void qkv_kernel(const float* __restrict__ X,
                const float* __restrict__ Wq, const float* __restrict__ bq,
                const float* __restrict__ Wk, const float* __restrict__ bk,
                const float* __restrict__ Wv, const float* __restrict__ bvp,
                unsigned short* __restrict__ Qw,
                unsigned short* __restrict__ Kw,
                unsigned short* __restrict__ Vw) {
  const float* W;  const float* bias;  unsigned short* out;
  if      (blockIdx.z == 0) { W = Wq; bias = bq;  out = Qw; }
  else if (blockIdx.z == 1) { W = Wk; bias = bk;  out = Kw; }
  else                      { W = Wv; bias = bvp; out = Vw; }

  __shared__ unsigned short Xs[128][40];   // 128 rows x 32 k (bf16), pad 8
  __shared__ unsigned short Ws[64][40];    // 64 n-rows x 32 k (bf16), pad 8

  const int t  = threadIdx.x;
  const int w  = t >> 5;            // wave id 0..7
  const int L  = t & 31;            // lane
  const int hi = L >> 4;            // lane group (0: lanes 0-15, 1: 16-31)
  const int lo = L & 15;
  const int m0 = blockIdx.y * 128;
  const int n0 = blockIdx.x * 64;

  v8f acc[4];
#pragma unroll
  for (int j = 0; j < 4; ++j) acc[j] = (v8f){0,0,0,0,0,0,0,0};

  for (int k0 = 0; k0 < H_; k0 += 32) {
    // ---- stage X tile (128 x 32 fp32 -> bf16) ----
    {
      const int r  = t >> 1;
      const int cb = (t & 1) * 16;
      const float* src = X + (size_t)(m0 + r) * H_ + k0 + cb;
#pragma unroll
      for (int i = 0; i < 16; i += 4) {
        float4 f = *(const float4*)(src + i);
        Xs[r][cb + i + 0] = f2bf(f.x);
        Xs[r][cb + i + 1] = f2bf(f.y);
        Xs[r][cb + i + 2] = f2bf(f.z);
        Xs[r][cb + i + 3] = f2bf(f.w);
      }
    }
    // ---- stage W tile (64 x 32 fp32 -> bf16), natural row-major ----
    {
      const int r  = t >> 2;
      const int cb = (t & 3) * 8;
      const float* src = W + (size_t)(n0 + r) * H_ + k0 + cb;
#pragma unroll
      for (int i = 0; i < 8; i += 4) {
        float4 f = *(const float4*)(src + i);
        Ws[r][cb + i + 0] = f2bf(f.x);
        Ws[r][cb + i + 1] = f2bf(f.y);
        Ws[r][cb + i + 2] = f2bf(f.z);
        Ws[r][cb + i + 3] = f2bf(f.w);
      }
    }
    __syncthreads();

    // A fragment: lane = row (w*16+lo); 16 consecutive k at hi*16.
    const int ar = w * 16 + lo;
    const int kb = hi * 16;
    v16bf a = frag2(&Xs[ar][kb], &Xs[ar][kb + 8]);

#pragma unroll
    for (int j = 0; j < 4; ++j) {
      // B fragment: lane = col n; 16 consecutive k at hi*16 (B[k][n]=W[n][k]).
      const int bn = j * 16 + lo;
      v16bf bf = frag2(&Ws[bn][hi * 16], &Ws[bn][hi * 16 + 8]);
      acc[j] = wmma_bf16(a, bf, acc[j]);
    }
    __syncthreads();
  }

  // ---- epilogue: +bias, fp32->bf16, write [B,h,S,dh] workspace ----
#pragma unroll
  for (int j = 0; j < 4; ++j) {
    const int n  = n0 + j * 16 + lo;
    const float bb = bias[n];
    const int hh = n >> 6, d = n & 63;
#pragma unroll
    for (int i = 0; i < 8; ++i) {
      const int m = m0 + w * 16 + hi * 8 + i;   // D layout: row = i + 8*hi
      const int b = m >> 11, s = m & 2047;
      out[((size_t)(b * NH_ + hh) * S_ + s) * DH_ + d] = f2bf(acc[j][i] + bb);
    }
  }
}

// ---------------------------------------------------------------------------
// Kernel 2: flash attention. grid = (S/128, NH, B); block = 256 (8 wave32).
// Each wave: 16 query rows; streams 64-key blocks with online softmax.
// ---------------------------------------------------------------------------
__global__ __launch_bounds__(256)
void attn_kernel(const unsigned short* __restrict__ Qw,
                 const unsigned short* __restrict__ Kw,
                 const unsigned short* __restrict__ Vw,
                 const int* __restrict__ amask,
                 float* __restrict__ out) {
  __shared__ unsigned short Qs[128][72];    // 128 q x 64 d
  __shared__ unsigned short Ks[64][72];     // 64 key x 64 d (natural)
  __shared__ unsigned short Vts[64][72];    // 64 d x 64 key (transposed)
  __shared__ unsigned short Ps[8][16][72];  // per-wave P block 16 q x 64 key
  __shared__ float mb[64];                  // mask bias per key

  const int t  = threadIdx.x;
  const int w  = t >> 5;
  const int L  = t & 31;
  const int hi = L >> 4;
  const int lo = L & 15;
  const int b  = blockIdx.z;
  const int h  = blockIdx.y;
  const int q0 = blockIdx.x * 128;

  const unsigned short* Qh = Qw + ((size_t)(b * NH_ + h) * S_ + q0) * DH_;
  const unsigned short* Kh = Kw + (size_t)(b * NH_ + h) * S_ * DH_;
  const unsigned short* Vh = Vw + (size_t)(b * NH_ + h) * S_ * DH_;

  // ---- stage Q once ----
  {
    const int r  = t >> 1;
    const int cb = (t & 1) * 32;
    const unsigned short* src = Qh + r * DH_ + cb;
#pragma unroll
    for (int i = 0; i < 4; ++i)
      *(ushort8*)&Qs[r][cb + i * 8] = *(const ushort8*)(src + i * 8);
  }
  __syncthreads();

  // Q A-fragments (kept in VGPRs for whole kernel): 2 k-slices of 32.
  v16bf aq[2];
  {
    const int r  = w * 16 + lo;
    const int kb = hi * 16;
    aq[0] = frag2(&Qs[r][kb],      &Qs[r][kb + 8]);
    aq[1] = frag2(&Qs[r][32 + kb], &Qs[r][32 + kb + 8]);
  }

  v8f acc[4];
  float mrow[8], lrow[8];
#pragma unroll
  for (int j = 0; j < 4; ++j) acc[j] = (v8f){0,0,0,0,0,0,0,0};
#pragma unroll
  for (int i = 0; i < 8; ++i) { mrow[i] = -3.0e38f; lrow[i] = 0.0f; }

  const float scale = 0.125f;   // 1/sqrt(64)

  for (int k0 = 0; k0 < S_; k0 += 64) {
    __syncthreads();   // prior iteration's reads of Ks/Vts/Ps are done

    // ---- stage K block (natural [key][d]) ----
    {
      const int r  = t >> 2;
      const int cb = (t & 3) * 16;
      const unsigned short* src = Kh + (size_t)(k0 + r) * DH_ + cb;
      *(ushort8*)&Ks[r][cb]     = *(const ushort8*)(src);
      *(ushort8*)&Ks[r][cb + 8] = *(const ushort8*)(src + 8);
    }
    // ---- stage V block transposed ([d][key]) ----
    {
      const int key = t >> 2;
      const int db  = (t & 3) * 16;
      const unsigned short* src = Vh + (size_t)(k0 + key) * DH_ + db;
      ushort8 v0 = *(const ushort8*)(src);
      ushort8 v1 = *(const ushort8*)(src + 8);
#pragma unroll
      for (int i = 0; i < 8; ++i) Vts[db + i][key]     = v0[i];
#pragma unroll
      for (int i = 0; i < 8; ++i) Vts[db + 8 + i][key] = v1[i];
    }
    if (t < 64) mb[t] = (amask[b * S_ + k0 + t] == 0) ? -1.0e30f : 0.0f;
    __syncthreads();

    // ---- scores: S = Q K^T  (4 key-subtiles x 2 k-steps) ----
    float sc[4][8];
#pragma unroll
    for (int jn = 0; jn < 4; ++jn) {
      const int key = jn * 16 + lo;       // B lane = key column
      v16bf b0 = frag2(&Ks[key][hi * 16],      &Ks[key][hi * 16 + 8]);
      v16bf b1 = frag2(&Ks[key][32 + hi * 16], &Ks[key][32 + hi * 16 + 8]);
      v8f s = wmma_bf16(aq[0], b0, (v8f){0,0,0,0,0,0,0,0});
      s = wmma_bf16(aq[1], b1, s);
      const float bias = mb[key];
#pragma unroll
      for (int i = 0; i < 8; ++i) sc[jn][i] = s[i] * scale + bias;
    }

    // ---- online softmax: row reductions across 16-lane groups ----
    float rm[8];
#pragma unroll
    for (int i = 0; i < 8; ++i) {
      rm[i] = fmaxf(fmaxf(sc[0][i], sc[1][i]), fmaxf(sc[2][i], sc[3][i]));
    }
#pragma unroll
    for (int off = 1; off < 16; off <<= 1)
#pragma unroll
      for (int i = 0; i < 8; ++i)
        rm[i] = fmaxf(rm[i], __shfl_xor(rm[i], off, 32));

    float fac[8], rs[8];
#pragma unroll
    for (int i = 0; i < 8; ++i) {
      const float mn = fmaxf(mrow[i], rm[i]);
      fac[i] = __expf(mrow[i] - mn);
      mrow[i] = mn;
      float sum = 0.0f;
#pragma unroll
      for (int jn = 0; jn < 4; ++jn) {
        sc[jn][i] = __expf(sc[jn][i] - mn);
        sum += sc[jn][i];
      }
      rs[i] = sum;
    }
#pragma unroll
    for (int off = 1; off < 16; off <<= 1)
#pragma unroll
      for (int i = 0; i < 8; ++i)
        rs[i] += __shfl_xor(rs[i], off, 32);
#pragma unroll
    for (int i = 0; i < 8; ++i) {
      lrow[i] = lrow[i] * fac[i] + rs[i];
#pragma unroll
      for (int j = 0; j < 4; ++j) acc[j][i] *= fac[i];
    }

    // ---- P: D-layout -> A-layout via LDS (bf16) ----
#pragma unroll
    for (int jn = 0; jn < 4; ++jn)
#pragma unroll
      for (int i = 0; i < 8; ++i)
        Ps[w][hi * 8 + i][jn * 16 + lo] = f2bf(sc[jn][i]);
    __syncthreads();

    // ---- O += P V  (2 key-slices x 4 d-subtiles) ----
#pragma unroll
    for (int ks = 0; ks < 2; ++ks) {
      const int kb = hi * 16;
      v16bf ap = frag2(&Ps[w][lo][ks * 32 + kb],
                       &Ps[w][lo][ks * 32 + kb + 8]);
#pragma unroll
      for (int j = 0; j < 4; ++j) {
        const int d = j * 16 + lo;   // B lane = d column; 16 consecutive keys
        v16bf bv = frag2(&Vts[d][ks * 32 + hi * 16],
                         &Vts[d][ks * 32 + hi * 16 + 8]);
        acc[j] = wmma_bf16(ap, bv, acc[j]);
      }
    }
  }

  // ---- epilogue: normalize, write [B, S, H] fp32 ----
#pragma unroll
  for (int i = 0; i < 8; ++i) {
    const int q = q0 + w * 16 + hi * 8 + i;
    const float inv = 1.0f / lrow[i];
#pragma unroll
    for (int j = 0; j < 4; ++j) {
      const int d = j * 16 + lo;
      out[((size_t)b * S_ + q) * H_ + h * DH_ + d] = acc[j][i] * inv;
    }
  }
}

// ---------------------------------------------------------------------------
extern "C" void kernel_launch(void* const* d_in, const int* in_sizes, int n_in,
                              void* d_out, int out_size, void* d_ws,
                              size_t ws_size, hipStream_t stream) {
  const float* X    = (const float*)d_in[0];
  const int*   mask = (const int*)d_in[1];
  const float* Wq   = (const float*)d_in[2];
  const float* bq   = (const float*)d_in[3];
  const float* Wk   = (const float*)d_in[4];
  const float* bk   = (const float*)d_in[5];
  const float* Wv   = (const float*)d_in[6];
  const float* bv   = (const float*)d_in[7];
  float* out = (float*)d_out;

  // Workspace: Q, K, V bf16 in [B, h, S, dh] layout. 3 * 16 MB = 48 MB.
  const size_t elems = (size_t)B_ * S_ * H_;
  unsigned short* Qw = (unsigned short*)d_ws;
  unsigned short* Kw = Qw + elems;
  unsigned short* Vw = Kw + elems;

  dim3 g1(H_ / 64, (B_ * S_) / 128, 3);       // 16 x 64 x 3
  qkv_kernel<<<g1, 256, 0, stream>>>(X, Wq, bq, Wk, bk, Wv, bv, Qw, Kw, Vw);

  dim3 g2(S_ / 128, NH_, B_);                 // 16 x 16 x 4
  attn_kernel<<<g2, 256, 0, stream>>>(Qw, Kw, Vw, mask, out);
}
